// TransformerLayer_29729763623353
// MI455X (gfx1250) — matrix operand order repaired and checked by reading, not compiled, not run on hardware
//
#include <hip/hip_runtime.h>
#include <hip/hip_bf16.h>
#include <math.h>

// ---------------------------------------------------------------------------
// MI455X (gfx1250) transformer layer.
// All matmuls on v_wmma_f32_16x16x32_bf16 (wave32), fp32 accumulate.
// LDS tiles pre-swizzled into WMMA fragment layout (2x ds_load_b128/fragment).
// Attention is fused flash-style: scores never touch HBM.
// ---------------------------------------------------------------------------

typedef __attribute__((ext_vector_type(16))) __bf16    v16bf;
typedef __attribute__((ext_vector_type(8)))  float     v8f;
typedef __attribute__((ext_vector_type(4)))  unsigned  u32x4;
typedef __attribute__((ext_vector_type(8)))  unsigned  u32x8;

#define BB  4
#define NN  512
#define MM  2048
#define DD  1024
#define HH  16
#define HD  64
#define FFD 4096

static __device__ __forceinline__ unsigned short f2bfu(float f) {
  unsigned u = __builtin_bit_cast(unsigned, f);
  unsigned r = (u + 0x7FFFu + ((u >> 16) & 1u)) >> 16;
  return (unsigned short)r;
}
static __device__ __forceinline__ unsigned pack2(float x, float y) {
  return (unsigned)f2bfu(x) | ((unsigned)f2bfu(y) << 16);
}

static __device__ __forceinline__ float gelu_tanh(float x) {
  float x3 = x * x * x;
  return 0.5f * x * (1.0f + tanhf(0.7978845608028654f * (x + 0.044715f * x3)));
}

// Two 16B chunks per lane -> two ds_load_b128.
static __device__ __forceinline__ v16bf load_frag(const unsigned* Sh, int byteBase) {
  u32x4 lo = *(const u32x4*)((const char*)Sh + byteBase);
  u32x4 hi = *(const u32x4*)((const char*)Sh + byteBase + 512);
  u32x8 c  = __builtin_shufflevector(lo, hi, 0, 1, 2, 3, 4, 5, 6, 7);
  return __builtin_bit_cast(v16bf, c);
}

// A element (r,c), r in [0,16) per 16-row strip, c in [0,32):
//   lane = ((c>>3)&1)*16 + r ; h = (c&7) + ((c>>4)<<3)
static __device__ __forceinline__ int a_off(int r, int c) {
  return ((r >> 4) << 10) + ((c >> 4) << 9) +
         ((((c >> 3) & 1) << 4) + (r & 15)) * 16 + ((c & 7) << 1);
}
// B element (k,n), k in [0,32), n tiles of 16: lane = ((k>>3)&1)*16 + (n&15), h = (k&7) + ((k>>4)<<3)
static __device__ __forceinline__ int b_off(int k, int n) {
  return ((n >> 4) << 10) + ((k >> 4) << 9) +
         ((((k >> 3) & 1) << 4) + (n & 15)) * 16 + ((k & 7) << 1);
}

// ---------------- wave32 reductions ----------------
static __device__ __forceinline__ float half_max(float v) {   // within 16-lane half
  v = fmaxf(v, __shfl_xor(v, 8, 32)); v = fmaxf(v, __shfl_xor(v, 4, 32));
  v = fmaxf(v, __shfl_xor(v, 2, 32)); v = fmaxf(v, __shfl_xor(v, 1, 32));
  return v;
}
static __device__ __forceinline__ float half_sum(float v) {
  v += __shfl_xor(v, 8, 32); v += __shfl_xor(v, 4, 32);
  v += __shfl_xor(v, 2, 32); v += __shfl_xor(v, 1, 32);
  return v;
}
static __device__ __forceinline__ float wave_sum(float v) {
  v += __shfl_xor(v, 16, 32);
  return half_sum(v);
}
static __device__ __forceinline__ float wave_max(float v) {
  v = fmaxf(v, __shfl_xor(v, 16, 32));
  return half_max(v);
}
template<bool MAX>
static __device__ __forceinline__ float block_red(float v, float* sm) {
  v = MAX ? wave_max(v) : wave_sum(v);
  const int lane = threadIdx.x & 31, w = threadIdx.x >> 5;
  if (lane == 0) sm[w] = v;
  __syncthreads();
  float r = sm[0];
  #pragma unroll
  for (int i = 1; i < 8; ++i) r = MAX ? fmaxf(r, sm[i]) : (r + sm[i]);
  __syncthreads();
  return r;
}

// ---------------------------------------------------------------------------
// WMMA GEMM. OUT[r,c] = epi( sum_k A[r,k]*B[k,c] + bias[c] )
// BM x BN block, BK=32, 256 threads = 8 waves (4 row x 2 col strips).
// EPI: 0 bias, 1 gelu(bias+acc), 2 res + (bias+acc)*ls
// ---------------------------------------------------------------------------
template<int BM, int BN, int EPI>
static __device__ __forceinline__ void gemm_tile(
    const float* __restrict__ A, int lda,
    const float* __restrict__ Bmat, int ldb,
    const float* __restrict__ bias,
    float* __restrict__ OUT, int ldo,
    const float* __restrict__ res,
    const float* __restrict__ ls,
    int K)
{
  constexpr int BK = 32;
  constexpr int NT = BN / 32;
  constexpr int AW = (BM / 16) * 256;
  constexpr int BW = (BN / 16) * 256;

  __shared__ unsigned As[2 * AW];
  __shared__ unsigned Bs[2 * BW];

  const int tid  = threadIdx.x;
  const int lane = tid & 31;
  const int wid  = tid >> 5;
  const int wr   = wid >> 1;
  const int wc   = wid & 1;

  const int rowBase = blockIdx.y * BM;
  const int colBase = blockIdx.x * BN;

  v8f acc[NT];
  #pragma unroll
  for (int t = 0; t < NT; ++t)
    #pragma unroll
    for (int r = 0; r < 8; ++r) acc[t][r] = 0.0f;

  auto stageA = [&](int k0, int buf) {
    unsigned* dst = As + buf * AW;
    constexpr int GROUPS = (BM * BK) / 4;
    for (int i = tid; i < GROUPS; i += 256) {
      int r = i >> 3;
      int c = (i & 7) << 2;
      float4 f = *(const float4*)(A + (size_t)(rowBase + r) * lda + (k0 + c));
      unsigned* q = (unsigned*)((char*)dst + a_off(r, c));
      q[0] = pack2(f.x, f.y);
      q[1] = pack2(f.z, f.w);
    }
  };
  auto stageB = [&](int k0, int buf) {
    unsigned* dst = Bs + buf * BW;
    constexpr int GROUPS = (BK / 2) * (BN / 4);
    for (int i = tid; i < GROUPS; i += 256) {
      int kk = (i / (BN / 4)) * 2;
      int n4 = (i % (BN / 4)) * 4;
      float4 f0 = *(const float4*)(Bmat + (size_t)(k0 + kk)     * ldb + colBase + n4);
      float4 f1 = *(const float4*)(Bmat + (size_t)(k0 + kk + 1) * ldb + colBase + n4);
      float a0[4] = {f0.x, f0.y, f0.z, f0.w};
      float a1[4] = {f1.x, f1.y, f1.z, f1.w};
      #pragma unroll
      for (int j = 0; j < 4; ++j)
        *(unsigned*)((char*)dst + b_off(kk, n4 + j)) = pack2(a0[j], a1[j]);
    }
  };

  const int KSteps = K / BK;
  stageA(0, 0);
  stageB(0, 0);
  __syncthreads();

  for (int ks = 0; ks < KSteps; ++ks) {
    const int cur = ks & 1;
    if (ks + 1 < KSteps) {
      stageA((ks + 1) * BK, cur ^ 1);
      stageB((ks + 1) * BK, cur ^ 1);
    }
    v16bf a = load_frag(As + cur * AW, (wr << 10) + lane * 16);
    #pragma unroll
    for (int t = 0; t < NT; ++t) {
      int st = wc * NT + t;
      v16bf b = load_frag(Bs + cur * BW, (st << 10) + lane * 16);
      acc[t] = __builtin_amdgcn_wmma_f32_16x16x32_bf16(
          false, a, false, b, (short)0, acc[t], false, false);
    }
    __syncthreads();
  }

  const int rowW  = rowBase + wr * 16 + ((lane >> 4) << 3);
  const int colW0 = colBase + wc * (BN / 2);
  const int cl    = lane & 15;
  #pragma unroll
  for (int t = 0; t < NT; ++t) {
    int col  = colW0 + t * 16 + cl;
    float bv = bias ? bias[col] : 0.0f;
    #pragma unroll
    for (int r = 0; r < 8; ++r) {
      size_t idx = (size_t)(rowW + r) * ldo + col;
      float x = acc[t][r] + bv;
      if (EPI == 1)      x = gelu_tanh(x);
      else if (EPI == 2) x = res[idx] + x * ls[col];
      OUT[idx] = x;
    }
  }
}

template<int BM, int BN, int EPI>
__global__ __launch_bounds__(256)
void wmma_gemm_kernel(const float* __restrict__ A, int lda,
                      const float* __restrict__ Bmat, int ldb,
                      const float* __restrict__ bias,
                      float* __restrict__ OUT, int ldo,
                      const float* __restrict__ res,
                      const float* __restrict__ ls,
                      int K)
{
  gemm_tile<BM, BN, EPI>(A, lda, Bmat, ldb, bias, OUT, ldo, res, ls, K);
}

// ---------------------------------------------------------------------------
// Fused flash attention: O[b,n,h*64+c] = softmax_m(Q.K^T * hd^-0.5) @ V
// Block: 128 query rows x one (b,h); 8 waves, each owns 16 rows x 64 cols.
// Q staged once (pre-scaled). KV chunks of 64, double buffered.
// Online softmax in registers; P transposed to A-frag layout through
// wave-private LDS (same-wave DS ordering, no barrier needed).
// ---------------------------------------------------------------------------
__global__ __launch_bounds__(256)
void flash_attn_kernel(const float* __restrict__ Q, const float* __restrict__ Kc,
                       const float* __restrict__ Vc, float* __restrict__ O,
                       int Mlen)
{
  __shared__ unsigned lds[65536 / 4];          // 64KB
  unsigned* Qs = lds;                          // 16KB: 8 strips x 2 ksteps
  unsigned* Ks = lds + 16384 / 4;              // 2 x 8KB
  unsigned* Vs = lds + 32768 / 4;              // 2 x 8KB
  unsigned* Ps = lds + 49152 / 4;              // 16KB: 2KB per wave

  const int tid  = threadIdx.x;
  const int lane = tid & 31;
  const int w    = tid >> 5;
  const int z = blockIdx.z, b = z / HH, h = z % HH;
  const int qRow0 = blockIdx.y * 128;

  const float* Qg = Q  + (size_t)b * NN   * DD + (size_t)h * HD + (size_t)qRow0 * DD;
  const float* Kg = Kc + (size_t)b * Mlen * DD + (size_t)h * HD;
  const float* Vg = Vc + (size_t)b * Mlen * DD + (size_t)h * HD;

  // ---- stage Q (128 x 64), scaled by hd^-0.5, into A-fragment layout ----
  for (int i = tid; i < 2048; i += 256) {
    int r = i >> 4;                 // 16 float4 per row
    int c = (i & 15) << 2;
    float4 f = *(const float4*)(Qg + (size_t)r * DD + c);
    unsigned* q = (unsigned*)((char*)Qs + (r >> 4) * 2048 + (c >> 5) * 1024 +
                              a_off(r & 15, c & 31));
    q[0] = pack2(f.x * 0.125f, f.y * 0.125f);
    q[1] = pack2(f.z * 0.125f, f.w * 0.125f);
  }

  auto stageKV = [&](int m0, int buf) {
    // K chunk (64 m x 64 k) as B^T fragments: per-lane data contiguous in k
    unsigned* kd = Ks + buf * (8192 / 4);
    for (int i = tid; i < 1024; i += 256) {
      int n = i >> 4;               // m row in chunk
      int c = (i & 15) << 2;        // k
      float4 f = *(const float4*)(Kg + (size_t)(m0 + n) * DD + c);
      unsigned* q = (unsigned*)((char*)kd + (c >> 5) * 4096 + b_off(c & 31, n));
      q[0] = pack2(f.x, f.y);
      q[1] = pack2(f.z, f.w);
    }
    // V chunk (64 m x 64 c) as B fragments: pack (m, m+1) pairs into b32 stores
    unsigned* vd = Vs + buf * (8192 / 4);
    for (int i = tid; i < 512; i += 256) {
      int mm = (i >> 4) << 1;
      int c4 = (i & 15) << 2;
      float4 f0 = *(const float4*)(Vg + (size_t)(m0 + mm)     * DD + c4);
      float4 f1 = *(const float4*)(Vg + (size_t)(m0 + mm + 1) * DD + c4);
      float a0[4] = {f0.x, f0.y, f0.z, f0.w};
      float a1[4] = {f1.x, f1.y, f1.z, f1.w};
      #pragma unroll
      for (int j = 0; j < 4; ++j)
        *(unsigned*)((char*)vd + (mm >> 5) * 4096 + b_off(mm & 31, c4 + j)) =
            pack2(a0[j], a1[j]);
    }
  };

  float mrow[8], lrow[8];
  #pragma unroll
  for (int r = 0; r < 8; ++r) { mrow[r] = -3.402823466e38f; lrow[r] = 0.0f; }
  v8f accO[4];
  #pragma unroll
  for (int t = 0; t < 4; ++t)
    #pragma unroll
    for (int r = 0; r < 8; ++r) accO[t][r] = 0.0f;

  stageKV(0, 0);
  __syncthreads();

  const int nChunks = Mlen / 64;
  unsigned* pw = Ps + w * (2048 / 4);
  const int hi = lane >> 4;
  const int cl = lane & 15;

  for (int ch = 0; ch < nChunks; ++ch) {
    const int cur = ch & 1;
    if (ch + 1 < nChunks) stageKV((ch + 1) * 64, cur ^ 1);
    const unsigned* kd = Ks + cur * (8192 / 4);
    const unsigned* vd = Vs + cur * (8192 / 4);

    // ---- S = Qs . K^T  (16 x 64 per wave, 8 WMMAs) ----
    v8f s[4];
    #pragma unroll
    for (int t = 0; t < 4; ++t)
      #pragma unroll
      for (int r = 0; r < 8; ++r) s[t][r] = 0.0f;
    #pragma unroll
    for (int ks = 0; ks < 2; ++ks) {
      v16bf aq = load_frag(Qs, w * 2048 + ks * 1024 + lane * 16);
      #pragma unroll
      for (int t = 0; t < 4; ++t) {
        v16bf bk = load_frag(kd, ks * 4096 + t * 1024 + lane * 16);
        s[t] = __builtin_amdgcn_wmma_f32_16x16x32_bf16(
            false, aq, false, bk, (short)0, s[t], false, false);
      }
    }

    // ---- online softmax update (rows live in 16-lane halves) ----
    float alpha[8];
    #pragma unroll
    for (int r = 0; r < 8; ++r) {
      float mx = fmaxf(fmaxf(s[0][r], s[1][r]), fmaxf(s[2][r], s[3][r]));
      mx = half_max(mx);
      float mnew = fmaxf(mrow[r], mx);
      alpha[r] = __expf(mrow[r] - mnew);
      mrow[r] = mnew;
      float psum = 0.0f;
      #pragma unroll
      for (int t = 0; t < 4; ++t) {
        float p = __expf(s[t][r] - mnew);
        s[t][r] = p;
        psum += p;
      }
      lrow[r] = lrow[r] * alpha[r] + half_sum(psum);
    }

    // ---- P (bf16) -> wave-private LDS in A-fragment layout ----
    #pragma unroll
    for (int t = 0; t < 4; ++t) {
      int m = t * 16 + cl;
      #pragma unroll
      for (int r = 0; r < 8; ++r) {
        int qrow = r + (hi << 3);
        *(unsigned short*)((char*)pw + (m >> 5) * 1024 + a_off(qrow, m & 31)) =
            f2bfu(s[t][r]);
      }
    }

    // ---- rescale O, then O += P . V (8 WMMAs) ----
    #pragma unroll
    for (int t = 0; t < 4; ++t)
      #pragma unroll
      for (int r = 0; r < 8; ++r) accO[t][r] *= alpha[r];
    #pragma unroll
    for (int ks = 0; ks < 2; ++ks) {
      v16bf ap = load_frag(pw, ks * 1024 + lane * 16);
      #pragma unroll
      for (int t = 0; t < 4; ++t) {
        v16bf bv = load_frag(vd, ks * 4096 + t * 1024 + lane * 16);
        accO[t] = __builtin_amdgcn_wmma_f32_16x16x32_bf16(
            false, ap, false, bv, (short)0, accO[t], false, false);
      }
    }
    __syncthreads();
  }

  // ---- finalize: O / l, write out ----
  float* Og = O + (size_t)b * NN * DD + (size_t)h * HD + (size_t)qRow0 * DD;
  #pragma unroll
  for (int r = 0; r < 8; ++r) {
    float inv = 1.0f / lrow[r];
    int qrow = w * 16 + r + (hi << 3);
    #pragma unroll
    for (int t = 0; t < 4; ++t)
      Og[(size_t)qrow * DD + t * 16 + cl] = accO[t][r] * inv;
  }
}

__global__ __launch_bounds__(256)
void layernorm_kernel(const float* __restrict__ x, const float* __restrict__ g,
                      const float* __restrict__ bta, float* __restrict__ y)
{
  __shared__ float sm[8];
  const int tid = threadIdx.x;
  const float* xr = x + (size_t)blockIdx.x * DD;
  float vals[DD / 256];
  float s = 0.0f;
  #pragma unroll
  for (int i = 0; i < DD / 256; ++i) { vals[i] = xr[tid + i * 256]; s += vals[i]; }
  float mean = block_red<false>(s, sm) / (float)DD;
  float vs = 0.0f;
  #pragma unroll
  for (int i = 0; i < DD / 256; ++i) { float d = vals[i] - mean; vs += d * d; }
  float inv = rsqrtf(block_red<false>(vs, sm) / (float)DD + 1e-5f);
  float* yr = y + (size_t)blockIdx.x * DD;
  #pragma unroll
  for (int i = 0; i < DD / 256; ++i) {
    int c = tid + i * 256;
    yr[c] = (vals[i] - mean) * inv * g[c] + bta[c];
  }
}

__global__ __launch_bounds__(256)
void add2_kernel(const float* __restrict__ a, const float* __restrict__ b,
                 float* __restrict__ o, int n)
{
  int i = blockIdx.x * 256 + threadIdx.x;
  if (i < n) o[i] = a[i] + b[i];
}

// ---------------------------------------------------------------------------
extern "C" void kernel_launch(void* const* d_in, const int* in_sizes, int n_in,
                              void* d_out, int out_size, void* d_ws, size_t ws_size,
                              hipStream_t stream) {
  (void)in_sizes; (void)n_in; (void)out_size; (void)ws_size;
  const float* state   = (const float*)d_in[0];
  const float* patches = (const float*)d_in[1];
  const float* ppe     = (const float*)d_in[2];
  const float* squery  = (const float*)d_in[3];
  const float* ca_wq = (const float*)d_in[4];  const float* ca_bq = (const float*)d_in[5];
  const float* ca_wk = (const float*)d_in[6];  const float* ca_bk = (const float*)d_in[7];
  const float* ca_wv = (const float*)d_in[8];  const float* ca_bv = (const float*)d_in[9];
  const float* ca_wp = (const float*)d_in[10]; const float* ca_bp = (const float*)d_in[11];
  const float* sa_wq = (const float*)d_in[12]; const float* sa_bq = (const float*)d_in[13];
  const float* sa_wk = (const float*)d_in[14]; const float* sa_bk = (const float*)d_in[15];
  const float* sa_wv = (const float*)d_in[16]; const float* sa_bv = (const float*)d_in[17];
  const float* sa_wp = (const float*)d_in[18]; const float* sa_bp = (const float*)d_in[19];
  const float* w1 = (const float*)d_in[20];    const float* b1 = (const float*)d_in[21];
  const float* w2 = (const float*)d_in[22];    const float* b2 = (const float*)d_in[23];
  const float* n1_g = (const float*)d_in[24];  const float* n1_b = (const float*)d_in[25];
  const float* ls1  = (const float*)d_in[26];
  const float* n2_g = (const float*)d_in[27];  const float* n2_b = (const float*)d_in[28];
  const float* ls2  = (const float*)d_in[29];
  const float* n3_g = (const float*)d_in[30];  const float* n3_b = (const float*)d_in[31];
  const float* ls3  = (const float*)d_in[32];
  float* out = (float*)d_out;

  const size_t SZ_ST = (size_t)BB * NN * DD;
  const size_t SZ_KV = (size_t)BB * MM * DD;
  float* ws  = (float*)d_ws;
  float* stA = ws;            ws += SZ_ST;
  float* stB = ws;            ws += SZ_ST;
  float* kv  = ws;            ws += SZ_KV;
  float* lnb = ws;            ws += SZ_ST;
  float* qb  = ws;            ws += SZ_ST;
  float* kb  = ws;            ws += SZ_KV;
  float* vb  = ws;            ws += SZ_KV;
  float* ab  = ws;            ws += SZ_ST;
  float* ffb = ws;            ws += (size_t)BB * NN * FFD;

  const dim3 t256(256);
  const int NTOK = BB * NN;   // 2048
  const int MTOK = BB * MM;   // 8192

  add2_kernel<<<dim3((NTOK * DD + 255) / 256), t256, 0, stream>>>(state, squery, stA, NTOK * DD);
  add2_kernel<<<dim3((MTOK * DD + 255) / 256), t256, 0, stream>>>(patches, ppe, kv, MTOK * DD);

  // -------- cross attention --------
  layernorm_kernel<<<dim3(NTOK), t256, 0, stream>>>(stA, n1_g, n1_b, lnb);
  wmma_gemm_kernel<64, 128, 0><<<dim3(DD / 128, NTOK / 64), t256, 0, stream>>>(
      lnb, DD, ca_wq, DD, ca_bq, qb, DD, nullptr, nullptr, DD);
  wmma_gemm_kernel<64, 128, 0><<<dim3(DD / 128, MTOK / 64), t256, 0, stream>>>(
      kv, DD, ca_wk, DD, ca_bk, kb, DD, nullptr, nullptr, DD);
  wmma_gemm_kernel<64, 128, 0><<<dim3(DD / 128, MTOK / 64), t256, 0, stream>>>(
      patches, DD, ca_wv, DD, ca_bv, vb, DD, nullptr, nullptr, DD);
  flash_attn_kernel<<<dim3(1, NN / 128, BB * HH), t256, 0, stream>>>(qb, kb, vb, ab, MM);
  wmma_gemm_kernel<64, 128, 2><<<dim3(DD / 128, NTOK / 64), t256, 0, stream>>>(
      ab, DD, ca_wp, DD, ca_bp, stB, DD, stA, ls1, DD);

  // -------- self attention --------
  layernorm_kernel<<<dim3(NTOK), t256, 0, stream>>>(stB, n2_g, n2_b, lnb);
  wmma_gemm_kernel<64, 128, 0><<<dim3(DD / 128, NTOK / 64), t256, 0, stream>>>(
      lnb, DD, sa_wq, DD, sa_bq, qb, DD, nullptr, nullptr, DD);
  wmma_gemm_kernel<64, 128, 0><<<dim3(DD / 128, NTOK / 64), t256, 0, stream>>>(
      lnb, DD, sa_wk, DD, sa_bk, kb, DD, nullptr, nullptr, DD);
  wmma_gemm_kernel<64, 128, 0><<<dim3(DD / 128, NTOK / 64), t256, 0, stream>>>(
      lnb, DD, sa_wv, DD, sa_bv, vb, DD, nullptr, nullptr, DD);
  flash_attn_kernel<<<dim3(1, NN / 128, BB * HH), t256, 0, stream>>>(qb, kb, vb, ab, NN);
  wmma_gemm_kernel<64, 128, 2><<<dim3(DD / 128, NTOK / 64), t256, 0, stream>>>(
      ab, DD, sa_wp, DD, sa_bp, stA, DD, stB, ls2, DD);

  // -------- MLP --------
  layernorm_kernel<<<dim3(NTOK), t256, 0, stream>>>(stA, n3_g, n3_b, lnb);
  wmma_gemm_kernel<64, 128, 1><<<dim3(FFD / 128, NTOK / 64), t256, 0, stream>>>(
      lnb, DD, w1, FFD, b1, ffb, FFD, nullptr, nullptr, DD);
  wmma_gemm_kernel<64, 128, 2><<<dim3(DD / 128, NTOK / 64), t256, 0, stream>>>(
      ffb, FFD, w2, DD, b2, out, DD, stA, ls3, FFD);
}
